// MoEAdapter_55379308314954
// MI455X (gfx1250) — compile-verified, matrix operand order checked by reading, not run on hardware
//
#include <hip/hip_runtime.h>
#include <hip/hip_bf16.h>
#include <stdint.h>

// ---------------- problem constants (match reference) ----------------
#define HDIM     2048
#define IDIM     2048
#define NEXP     8
#define RLORA    16
#define NTOK     8192          // 4 * 2048 tokens
#define SCALING  2.0f          // 32 / 16
#define AUX_COEF 0.01f
#define Z_COEF   0.001f

#define TILE_M 128
#define TILE_N 128
#define TILE_K 32

typedef __attribute__((ext_vector_type(16))) __bf16 v16bf;
typedef __attribute__((ext_vector_type(8)))  float  v8f;

static __device__ __forceinline__ v8f wmma_bf16(v16bf a, v16bf b, v8f c) {
  return __builtin_amdgcn_wmma_f32_16x16x32_bf16(
      false, a, false, b, (short)0, c, false, false);
}

// CDNA5 async global->LDS copy (16B per lane), tracked by ASYNCcnt.
// Per-lane global addresses => free gather. Low 32 bits of a generic
// shared pointer == LDS byte offset on gfx1250.
static __device__ __forceinline__ void async_ld_b128(const void* g, void* l) {
  const uint32_t lds = (uint32_t)(uintptr_t)l;
  asm volatile("global_load_async_to_lds_b128 %0, %1, off"
               :: "v"(lds), "v"(g) : "memory");
}
static __device__ __forceinline__ void wait_async0() {
  asm volatile("s_wait_asynccnt 0" ::: "memory");
}

// copy one 32B row-chunk (2 x b128) into a [128 x 32] bf16 tile
static __device__ __forceinline__ void fill_row_async(
    __bf16* lds_tile, const __bf16* grow, int row, int half) {
  const __bf16* g = grow + half * 16;
  __bf16* l = lds_tile + row * TILE_K + half * 16;
  async_ld_b128(g, l);
  async_ld_b128(g + 8, l + 8);
}

// ---------------------------------------------------------------------
// One-time fp32 -> bf16 conversion (x and the three big weight tensors)
// ---------------------------------------------------------------------
__global__ __launch_bounds__(256)
void cvt_bf16_kernel(const float* __restrict__ src, __bf16* __restrict__ dst) {
  const size_t i = ((size_t)blockIdx.x * blockDim.x + threadIdx.x) * 4;
  const float4 v = *(const float4*)&src[i];
  dst[i + 0] = (__bf16)v.x;
  dst[i + 1] = (__bf16)v.y;
  dst[i + 2] = (__bf16)v.z;
  dst[i + 3] = (__bf16)v.w;
}

// init routed-token lists (tok=0, gate=0) + per-expert counters
__global__ __launch_bounds__(256)
void init_lists_kernel(int* __restrict__ tok_list, float* __restrict__ gate_list,
                       int* __restrict__ counts) {
  const int i = blockIdx.x * 256 + threadIdx.x;
  tok_list[i] = 0;
  gate_list[i] = 0.0f;
  if (i < NEXP) counts[i] = 0;
}

__global__ __launch_bounds__(256)
void zero_out_kernel(float* __restrict__ out) {
  const size_t i = (size_t)blockIdx.x * blockDim.x + threadIdx.x;
  out[i] = 0.0f;
}

// ---------------------------------------------------------------------
// Router: one wave32 per token; emits per-token top-2 (ids + softmax
// weights) and lse^2 for the z-loss. No atomics -> deterministic.
// ---------------------------------------------------------------------
__global__ __launch_bounds__(256)
void router_kernel(const float* __restrict__ x, const float* __restrict__ wg,
                   int* __restrict__ e0a, int* __restrict__ e1a,
                   float* __restrict__ g0a, float* __restrict__ g1a,
                   float* __restrict__ lse2) {
  const int wave = threadIdx.x >> 5;
  const int lane = threadIdx.x & 31;
  const int t = blockIdx.x * 8 + wave;

  float p[NEXP];
#pragma unroll
  for (int e = 0; e < NEXP; ++e) p[e] = 0.0f;

  for (int j = 0; j < HDIM / 32; ++j) {
    const int k = lane + j * 32;
    const float xv = x[(size_t)t * HDIM + k];
#pragma unroll
    for (int e = 0; e < NEXP; ++e) p[e] += xv * wg[e * HDIM + k];
  }
#pragma unroll
  for (int e = 0; e < NEXP; ++e)
    for (int off = 16; off > 0; off >>= 1) p[e] += __shfl_xor(p[e], off, 32);

  if (lane == 0) {
    int i0 = 0;
#pragma unroll
    for (int e = 1; e < NEXP; ++e) if (p[e] > p[i0]) i0 = e;
    int i1 = (i0 == 0) ? 1 : 0;
#pragma unroll
    for (int e = 0; e < NEXP; ++e) if (e != i0 && p[e] > p[i1]) i1 = e;

    const float e1 = __expf(p[i1] - p[i0]);
    const float den = 1.0f + e1;
    e0a[t] = i0; e1a[t] = i1;
    g0a[t] = 1.0f / den; g1a[t] = e1 / den;

    float mx = p[0];
#pragma unroll
    for (int e = 1; e < NEXP; ++e) mx = fmaxf(mx, p[e]);
    float s = 0.0f;
#pragma unroll
    for (int e = 0; e < NEXP; ++e) s += __expf(p[e] - mx);
    const float lse = mx + __logf(s);
    lse2[t] = lse * lse;
  }
}

// build per-expert token lists (list order nondeterministic, but every
// per-token float result is order-independent -> output deterministic)
__global__ __launch_bounds__(256)
void scatter_kernel(const int* __restrict__ e0a, const int* __restrict__ e1a,
                    const float* __restrict__ g0a, const float* __restrict__ g1a,
                    int* __restrict__ tok_list, float* __restrict__ gate_list,
                    int* __restrict__ counts) {
  const int t = blockIdx.x * 256 + threadIdx.x;
  const int i0 = e0a[t], i1 = e1a[t];
  int p0 = atomicAdd(&counts[i0], 1);
  tok_list[(size_t)i0 * NTOK + p0] = t;
  gate_list[(size_t)i0 * NTOK + p0] = g0a[t];
  int p1 = atomicAdd(&counts[i1], 1);
  tok_list[(size_t)i1 * NTOK + p1] = t;
  gate_list[(size_t)i1 * NTOK + p1] = g1a[t];
}

// deterministic routing-loss reduction: fixed-order strided partials +
// fixed-shape tree reduction in one block
__global__ __launch_bounds__(256)
void loss_kernel(const int* __restrict__ e0a, const int* __restrict__ e1a,
                 const float* __restrict__ g0a, const float* __restrict__ g1a,
                 const float* __restrict__ lse2, const int* __restrict__ counts,
                 float* __restrict__ out, int out_size) {
  __shared__ float simp[256][NEXP];
  __shared__ float sz[256];
  const int tid = threadIdx.x;
  float imp[NEXP];
#pragma unroll
  for (int e = 0; e < NEXP; ++e) imp[e] = 0.0f;
  float z = 0.0f;
  for (int t = tid; t < NTOK; t += 256) {
    imp[e0a[t]] += g0a[t];
    imp[e1a[t]] += g1a[t];
    z += lse2[t];
  }
#pragma unroll
  for (int e = 0; e < NEXP; ++e) simp[tid][e] = imp[e];
  sz[tid] = z;
  __syncthreads();
  for (int s = 128; s > 0; s >>= 1) {
    if (tid < s) {
#pragma unroll
      for (int e = 0; e < NEXP; ++e) simp[tid][e] += simp[tid + s][e];
      sz[tid] += sz[tid + s];
    }
    __syncthreads();
  }
  if (tid == 0) {
    float dot = 0.0f;
#pragma unroll
    for (int e = 0; e < NEXP; ++e) dot += simp[0][e] * (float)counts[e];
    const float tt = (float)NTOK * (float)NTOK;
    const float lb = AUX_COEF * ((float)NEXP * dot / tt);
    const float zl = Z_COEF * (sz[0] / (float)NTOK);
    out[out_size - 1] = lb + zl;
  }
}

// ---------------------------------------------------------------------
// Rank-16 LoRA pre-projections
// ---------------------------------------------------------------------
__global__ __launch_bounds__(256)
void lora_x_kernel(const float* __restrict__ x,
                   const float* __restrict__ Ag, const float* __restrict__ Au,
                   float* __restrict__ og, float* __restrict__ ou) {
  const int wave = threadIdx.x >> 5;
  const int lane = threadIdx.x & 31;
  const int t = blockIdx.x * 8 + wave;
  const int r = lane & 15;
  const float* A = (lane < 16) ? Ag : Au;
  float s = 0.0f;
  for (int k = 0; k < HDIM; ++k)
    s += x[(size_t)t * HDIM + k] * A[r * HDIM + k];
  float* o = (lane < 16) ? og : ou;
  o[(size_t)t * RLORA + r] = s;
}

// xda[pos] = a[pos,:] @ A_down^T  (position-major)
__global__ __launch_bounds__(256)
void lora_a_kernel(const __bf16* __restrict__ a, const float* __restrict__ Ad,
                   const int* __restrict__ cnt_ptr, float* __restrict__ od) {
  const int wave = threadIdx.x >> 5;
  const int lane = threadIdx.x & 31;
  const int pos = blockIdx.x * 8 + wave;
  const int lim = (cnt_ptr[0] + TILE_M - 1) & ~(TILE_M - 1);
  if (pos >= lim) return;            // no barriers in this kernel
  if (lane < 16) {
    float s = 0.0f;
    for (int k = 0; k < IDIM; ++k)
      s += (float)a[(size_t)pos * IDIM + k] * Ad[lane * IDIM + k];
    od[(size_t)pos * RLORA + lane] = s;
  }
}

// ---------------------------------------------------------------------
// Routed fused gate+up GEMM: A rows gathered per token via per-lane
// async global->LDS; double-buffered LDS; 16 WMMA per K-step.
// ---------------------------------------------------------------------
__global__ __launch_bounds__(256)
void gu_gemm_kernel(const __bf16* __restrict__ xb,   // [T,H] bf16
                    const int* __restrict__ tok_list,// [T] tokens of expert
                    const int* __restrict__ cnt_ptr, // [1]
                    const __bf16* __restrict__ gwb,  // [I,H] bf16
                    const __bf16* __restrict__ uwb,  // [I,H] bf16
                    const float* __restrict__ gB,    // [I,R] fp32
                    const float* __restrict__ uB,    // [I,R] fp32
                    const float* __restrict__ xga,   // [T,R] token-major
                    const float* __restrict__ xua,   // [T,R]
                    __bf16* __restrict__ a_out) {    // [T,I] position-major
  const int m0 = blockIdx.y * TILE_M;
  if (m0 >= cnt_ptr[0]) return;       // uniform early-exit, before barriers

  __shared__ __bf16 sA [2][TILE_M][TILE_K];
  __shared__ __bf16 sBg[2][TILE_N][TILE_K];
  __shared__ __bf16 sBu[2][TILE_N][TILE_K];

  const int tid   = threadIdx.x;
  const int lane  = tid & 31;
  const int wave  = tid >> 5;
  const int waveM = wave & 3;
  const int waveN = wave >> 2;
  const int n0 = blockIdx.x * TILE_N;

  const int frow = tid >> 1;          // fill: 2 threads per row
  const int fhalf = tid & 1;
  const __bf16* aRow = xb + (size_t)tok_list[m0 + frow] * HDIM;  // gather base

  auto fill = [&](int b, int k0) {
    fill_row_async(&sA [b][0][0], aRow + k0,                          frow, fhalf);
    fill_row_async(&sBg[b][0][0], gwb + (size_t)(n0 + frow) * HDIM + k0 - (size_t)frow * HDIM + (size_t)frow * HDIM, frow, fhalf);
    fill_row_async(&sBu[b][0][0], uwb + (size_t)(n0 + frow) * HDIM + k0, frow, fhalf);
  };
  // (A row uses gathered base; B rows are plain strided)
  auto fillB = [&](int b, int k0) {
    fill_row_async(&sBg[b][0][0], gwb + (size_t)(n0 + frow) * HDIM + k0, frow, fhalf);
  };
  (void)fillB;

  v8f cg[2][4], cu[2][4];
#pragma unroll
  for (int a = 0; a < 2; ++a)
#pragma unroll
    for (int b = 0; b < 4; ++b) { cg[a][b] = (v8f){}; cu[a][b] = (v8f){}; }

  const int lrow = lane & 15;
  const int koff = (lane >> 4) << 4;

  fill(0, 0);
  wait_async0();
  __syncthreads();

  int cur = 0;
  for (int k0 = 0; k0 < HDIM; k0 += TILE_K) {
    const int nxt = cur ^ 1;
    if (k0 + TILE_K < HDIM) fill(nxt, k0 + TILE_K);

    const v16bf a0 = *(const v16bf*)&sA[cur][waveM * 32 +      lrow][koff];
    const v16bf a1 = *(const v16bf*)&sA[cur][waveM * 32 + 16 + lrow][koff];
    {
      const v16bf b0 = *(const v16bf*)&sBg[cur][waveN * 64 +  0 + lrow][koff];
      const v16bf b1 = *(const v16bf*)&sBg[cur][waveN * 64 + 16 + lrow][koff];
      const v16bf b2 = *(const v16bf*)&sBg[cur][waveN * 64 + 32 + lrow][koff];
      const v16bf b3 = *(const v16bf*)&sBg[cur][waveN * 64 + 48 + lrow][koff];
      cg[0][0] = wmma_bf16(a0, b0, cg[0][0]);
      cg[1][0] = wmma_bf16(a1, b0, cg[1][0]);
      cg[0][1] = wmma_bf16(a0, b1, cg[0][1]);
      cg[1][1] = wmma_bf16(a1, b1, cg[1][1]);
      cg[0][2] = wmma_bf16(a0, b2, cg[0][2]);
      cg[1][2] = wmma_bf16(a1, b2, cg[1][2]);
      cg[0][3] = wmma_bf16(a0, b3, cg[0][3]);
      cg[1][3] = wmma_bf16(a1, b3, cg[1][3]);
    }
    {
      const v16bf b0 = *(const v16bf*)&sBu[cur][waveN * 64 +  0 + lrow][koff];
      const v16bf b1 = *(const v16bf*)&sBu[cur][waveN * 64 + 16 + lrow][koff];
      const v16bf b2 = *(const v16bf*)&sBu[cur][waveN * 64 + 32 + lrow][koff];
      const v16bf b3 = *(const v16bf*)&sBu[cur][waveN * 64 + 48 + lrow][koff];
      cu[0][0] = wmma_bf16(a0, b0, cu[0][0]);
      cu[1][0] = wmma_bf16(a1, b0, cu[1][0]);
      cu[0][1] = wmma_bf16(a0, b1, cu[0][1]);
      cu[1][1] = wmma_bf16(a1, b1, cu[1][1]);
      cu[0][2] = wmma_bf16(a0, b2, cu[0][2]);
      cu[1][2] = wmma_bf16(a1, b2, cu[1][2]);
      cu[0][3] = wmma_bf16(a0, b3, cu[0][3]);
      cu[1][3] = wmma_bf16(a1, b3, cu[1][3]);
    }

    wait_async0();
    __syncthreads();
    cur = nxt;
  }

  // epilogue: LoRA rank-16 + SiLU(g)*u, bf16 store (position-major)
  const int mbase = m0 + waveM * 32;
  const int nbase = n0 + waveN * 64;
  const int mhalf = (lane >> 4) << 3;
  const int ncol  = lane & 15;
#pragma unroll
  for (int tm = 0; tm < 2; ++tm)
#pragma unroll
    for (int tn = 0; tn < 4; ++tn)
#pragma unroll
      for (int v = 0; v < 8; ++v) {
        const int p = mbase + tm * 16 + mhalf + v;     // position in list
        const int t = tok_list[p];                      // token id
        const int i = nbase + tn * 16 + ncol;
        float g = cg[tm][tn][v];
        float u = cu[tm][tn][v];
        float lg = 0.0f, lu = 0.0f;
#pragma unroll
        for (int r = 0; r < RLORA; ++r) {
          lg += xga[(size_t)t * RLORA + r] * gB[(size_t)i * RLORA + r];
          lu += xua[(size_t)t * RLORA + r] * uB[(size_t)i * RLORA + r];
        }
        g += SCALING * lg;
        u += SCALING * lu;
        const float s = g / (1.0f + __expf(-g));        // silu
        a_out[(size_t)p * IDIM + i] = (__bf16)(s * u);
      }
}

// ---------------------------------------------------------------------
// Routed down GEMM: contiguous position-major A; epilogue scatters
// gate-weighted rows to out[token]. Padded rows carry gate==0.
// ---------------------------------------------------------------------
__global__ __launch_bounds__(256)
void down_gemm_kernel(const __bf16* __restrict__ a,     // [T,I] position-major
                      const int* __restrict__ tok_list, // [T]
                      const float* __restrict__ gate_list,
                      const int* __restrict__ cnt_ptr,
                      const __bf16* __restrict__ dwb,   // [H,I] bf16
                      const float* __restrict__ dB,     // [H,R] fp32
                      const float* __restrict__ xda,    // [T,R] position-major
                      float* __restrict__ out) {        // [T,H] fp32 (zeroed)
  const int m0 = blockIdx.y * TILE_M;
  if (m0 >= cnt_ptr[0]) return;

  __shared__ __bf16 sA[2][TILE_M][TILE_K];
  __shared__ __bf16 sB[2][TILE_N][TILE_K];

  const int tid   = threadIdx.x;
  const int lane  = tid & 31;
  const int wave  = tid >> 5;
  const int waveM = wave & 3;
  const int waveN = wave >> 2;
  const int n0 = blockIdx.x * TILE_N;

  const int frow = tid >> 1;
  const int fhalf = tid & 1;

  auto fill = [&](int b, int k0) {
    fill_row_async(&sA[b][0][0], a   + (size_t)(m0 + frow) * IDIM + k0, frow, fhalf);
    fill_row_async(&sB[b][0][0], dwb + (size_t)(n0 + frow) * IDIM + k0, frow, fhalf);
  };

  v8f c[2][4];
#pragma unroll
  for (int x0 = 0; x0 < 2; ++x0)
#pragma unroll
    for (int y0 = 0; y0 < 4; ++y0) c[x0][y0] = (v8f){};

  const int lrow = lane & 15;
  const int koff = (lane >> 4) << 4;

  fill(0, 0);
  wait_async0();
  __syncthreads();

  int cur = 0;
  for (int k0 = 0; k0 < IDIM; k0 += TILE_K) {
    const int nxt = cur ^ 1;
    if (k0 + TILE_K < IDIM) fill(nxt, k0 + TILE_K);

    const v16bf a0 = *(const v16bf*)&sA[cur][waveM * 32 +      lrow][koff];
    const v16bf a1 = *(const v16bf*)&sA[cur][waveM * 32 + 16 + lrow][koff];
    const v16bf b0 = *(const v16bf*)&sB[cur][waveN * 64 +  0 + lrow][koff];
    const v16bf b1 = *(const v16bf*)&sB[cur][waveN * 64 + 16 + lrow][koff];
    const v16bf b2 = *(const v16bf*)&sB[cur][waveN * 64 + 32 + lrow][koff];
    const v16bf b3 = *(const v16bf*)&sB[cur][waveN * 64 + 48 + lrow][koff];
    c[0][0] = wmma_bf16(a0, b0, c[0][0]);
    c[1][0] = wmma_bf16(a1, b0, c[1][0]);
    c[0][1] = wmma_bf16(a0, b1, c[0][1]);
    c[1][1] = wmma_bf16(a1, b1, c[1][1]);
    c[0][2] = wmma_bf16(a0, b2, c[0][2]);
    c[1][2] = wmma_bf16(a1, b2, c[1][2]);
    c[0][3] = wmma_bf16(a0, b3, c[0][3]);
    c[1][3] = wmma_bf16(a1, b3, c[1][3]);

    wait_async0();
    __syncthreads();
    cur = nxt;
  }

  const int mbase = m0 + waveM * 32;
  const int nbase = n0 + waveN * 64;
  const int mhalf = (lane >> 4) << 3;
  const int ncol  = lane & 15;
#pragma unroll
  for (int tm = 0; tm < 2; ++tm)
#pragma unroll
    for (int tn = 0; tn < 4; ++tn)
#pragma unroll
      for (int v = 0; v < 8; ++v) {
        const int p = mbase + tm * 16 + mhalf + v;
        const int t = tok_list[p];
        const float gv = gate_list[p];
        const int h = nbase + tn * 16 + ncol;
        float o = c[tm][tn][v];
        float l = 0.0f;
#pragma unroll
        for (int r = 0; r < RLORA; ++r)
          l += xda[(size_t)p * RLORA + r] * dB[(size_t)h * RLORA + r];
        o += SCALING * l;
        out[(size_t)t * HDIM + h] += gv * o;   // one writer per (t,h) per launch
      }
}

// out = hidden + alpha * acc (acc lives in d_out already)
__global__ __launch_bounds__(256)
void final_kernel(const float* __restrict__ hidden,
                  const float* __restrict__ alpha, float* __restrict__ out) {
  const size_t i = (size_t)blockIdx.x * blockDim.x + threadIdx.x;
  out[i] = hidden[i] + alpha[0] * out[i];
}

// ---------------------------------------------------------------------
extern "C" void kernel_launch(void* const* d_in, const int* in_sizes, int n_in,
                              void* d_out, int out_size, void* d_ws, size_t ws_size,
                              hipStream_t stream) {
  (void)in_sizes; (void)n_in; (void)ws_size;
  const float* x      = (const float*)d_in[0];
  const float* w_gate = (const float*)d_in[1];
  const float* gate_w = (const float*)d_in[2];
  const float* gate_A = (const float*)d_in[3];
  const float* gate_B = (const float*)d_in[4];
  const float* up_w   = (const float*)d_in[5];
  const float* up_A   = (const float*)d_in[6];
  const float* up_B   = (const float*)d_in[7];
  const float* down_w = (const float*)d_in[8];
  const float* down_A = (const float*)d_in[9];
  const float* down_B = (const float*)d_in[10];
  const float* alpha  = (const float*)d_in[11];
  float* out = (float*)d_out;                     // [T*H] output + [1] loss

  // workspace carving
  uintptr_t base = (uintptr_t)d_ws;
  int*    counts   = (int*)base;    base += 256;
  int*    e0a      = (int*)base;    base += (size_t)NTOK * 4;
  int*    e1a      = (int*)base;    base += (size_t)NTOK * 4;
  float*  g0a      = (float*)base;  base += (size_t)NTOK * 4;
  float*  g1a      = (float*)base;  base += (size_t)NTOK * 4;
  float*  lse2     = (float*)base;  base += (size_t)NTOK * 4;
  int*    tok_list = (int*)base;    base += (size_t)NEXP * NTOK * 4;
  float*  gate_lst = (float*)base;  base += (size_t)NEXP * NTOK * 4;
  float*  xga      = (float*)base;  base += (size_t)NTOK * RLORA * 4;
  float*  xua      = (float*)base;  base += (size_t)NTOK * RLORA * 4;
  float*  xda      = (float*)base;  base += (size_t)NTOK * RLORA * 4;
  __bf16* a_buf    = (__bf16*)base; base += (size_t)NTOK * IDIM * 2;
  __bf16* xb       = (__bf16*)base; base += (size_t)NTOK * HDIM * 2;
  __bf16* gwb      = (__bf16*)base; base += (size_t)NEXP * IDIM * HDIM * 2;
  __bf16* uwb      = (__bf16*)base; base += (size_t)NEXP * IDIM * HDIM * 2;
  __bf16* dwb      = (__bf16*)base; base += (size_t)NEXP * HDIM * IDIM * 2;

  // one-time bf16 conversions
  const size_t NW = (size_t)NEXP * IDIM * HDIM;
  cvt_bf16_kernel<<<(NTOK * (size_t)HDIM) / 1024, 256, 0, stream>>>(x, xb);
  cvt_bf16_kernel<<<NW / 1024, 256, 0, stream>>>(gate_w, gwb);
  cvt_bf16_kernel<<<NW / 1024, 256, 0, stream>>>(up_w, uwb);
  cvt_bf16_kernel<<<NW / 1024, 256, 0, stream>>>(down_w, dwb);

  // routing
  init_lists_kernel<<<(NEXP * NTOK) / 256, 256, 0, stream>>>(tok_list, gate_lst, counts);
  zero_out_kernel<<<(NTOK * (size_t)HDIM) / 256, 256, 0, stream>>>(out);
  router_kernel<<<NTOK / 8, 256, 0, stream>>>(x, w_gate, e0a, e1a, g0a, g1a, lse2);
  scatter_kernel<<<NTOK / 256, 256, 0, stream>>>(e0a, e1a, g0a, g1a,
                                                 tok_list, gate_lst, counts);
  loss_kernel<<<1, 256, 0, stream>>>(e0a, e1a, g0a, g1a, lse2, counts, out, out_size);

  const dim3 ggrid(IDIM / TILE_N, NTOK / TILE_M);   // 16 x 64 (early-exit on count)
  const dim3 dgrid(HDIM / TILE_N, NTOK / TILE_M);

  for (int e = 0; e < NEXP; ++e) {
    int* tl = tok_list + (size_t)e * NTOK;
    float* gl = gate_lst + (size_t)e * NTOK;
    int* cp = counts + e;
    lora_x_kernel<<<NTOK / 8, 256, 0, stream>>>(
        x, gate_A + (size_t)e * RLORA * HDIM, up_A + (size_t)e * RLORA * HDIM,
        xga, xua);
    gu_gemm_kernel<<<ggrid, 256, 0, stream>>>(
        xb, tl, cp,
        gwb + (size_t)e * IDIM * HDIM, uwb + (size_t)e * IDIM * HDIM,
        gate_B + (size_t)e * IDIM * RLORA, up_B + (size_t)e * IDIM * RLORA,
        xga, xua, a_buf);
    lora_a_kernel<<<NTOK / 8, 256, 0, stream>>>(
        a_buf, down_A + (size_t)e * RLORA * IDIM, cp, xda);
    down_gemm_kernel<<<dgrid, 256, 0, stream>>>(
        a_buf, tl, gl, cp, dwb + (size_t)e * HDIM * IDIM,
        down_B + (size_t)e * HDIM * RLORA, xda, out);
  }

  final_kernel<<<(NTOK * HDIM) / 256, 256, 0, stream>>>(x, alpha, out);
}
